// SocialLstm_89481348645534
// MI455X (gfx1250) — matrix operand order repaired and checked by reading, not compiled, not run on hardware
//
#include <hip/hip_runtime.h>
#include <math.h>

// ---------------- problem constants (from reference) ----------------
#define T_TOT  24
#define NA     2048
#define T_OBS  9
#define T_PRED 19
#define HID    64
#define MED    128
#define SOC    64
#define ZK     256          // fused K: r(128) | e(64) | h(64)
#define G4     256          // 4*HID gates
#define CELLSZ 0.3f
#define AGB    16           // agents per block in fused step kernel

typedef _Float16 f16_t;
typedef __attribute__((ext_vector_type(16))) _Float16 v16h;
typedef __attribute__((ext_vector_type(8)))  float    v8f;

union Frag16 { v16h v; unsigned int u[8]; };

// A fragment: 16x32 f16, A row-major [*, lda], contiguous k order (k = 16*(lane/16) + s) (works for LDS or global ptrs)
__device__ __forceinline__ v16h load_a_frag(const f16_t* A, int lda, int row0, int k0, int lane) {
    Frag16 f;
    const unsigned int* p = (const unsigned int*)(A + (row0 + (lane & 15)) * lda + k0);
    const int q = (lane >> 4) * 8;      // lanes 0-15: K 0-15 ; lanes 16-31: K 16-31
#pragma unroll
    for (int j = 0; j < 4; ++j) {
        f.u[j]     = p[q + j];
        f.u[4 + j] = p[q + 4 + j];
    }
    return f.v;
}

// B fragment: 32x16 f16 where B[k][n] = W[n][k], W row-major [Nout, ldb(=K)]
__device__ __forceinline__ v16h load_b_frag(const f16_t* W, int ldb, int n0, int k0, int lane) {
    Frag16 f;
    const unsigned int* p =
        (const unsigned int*)(W + (n0 + (lane & 15)) * ldb + k0 + ((lane >> 4) << 4));
#pragma unroll
    for (int j = 0; j < 8; ++j) f.u[j] = p[j];
    return f.v;
}

// ---------------- one-time prep ----------------
__global__ __launch_bounds__(256)
void k_prep(const float* __restrict__ W_ih, const float* __restrict__ W_hh,
            const float* __restrict__ W_soc,
            const float* __restrict__ b_ih, const float* __restrict__ b_hh,
            const float* __restrict__ h0, const float* __restrict__ c0,
            f16_t* __restrict__ Wg_h, f16_t* __restrict__ Wsoc_h, float* __restrict__ b_g,
            float* __restrict__ h, float* __restrict__ c, float* __restrict__ out) {
    int tid = blockIdx.x * 256 + threadIdx.x;          // grid covers 131072
    if (tid < 256 * 256) {                             // fused gate weight [Wih | Whh], f16
        int n = tid >> 8, k = tid & 255;
        float w = (k < 192) ? W_ih[n * 192 + k] : W_hh[n * 64 + (k - 192)];
        Wg_h[tid] = (f16_t)w;
    }
    if (tid < 64 * 64) {                               // collapse 9 tiled blocks of W_soc
        int n = tid >> 6, k = tid & 63;
        float s = 0.f;
#pragma unroll
        for (int w = 0; w < 9; ++w) s += W_soc[n * 576 + w * 64 + k];
        Wsoc_h[tid] = (f16_t)s;
    }
    if (tid < 256) b_g[tid] = b_ih[tid] + b_hh[tid];
    if (tid < NA * HID) { h[tid] = h0[tid]; c[tid] = c0[tid]; }
    if (tid < 4 * NA * 2) out[20 * NA * 2 + tid] = 0.f;   // zero tail frames 20..23
}

// ---------------- per-frame grid cells: p = floor((x - (min-1.2))/0.3) * mask ----------------
__global__ __launch_bounds__(256)
void k_cells(const float* __restrict__ X, const float* __restrict__ masks,
             int2* __restrict__ cells) {
    int t = blockIdx.x, tid = threadIdx.x;
    __shared__ float sx[256], sy[256];
    const float* xt = X + t * NA * 2;
    float mx = 1e30f, my = 1e30f;
    for (int i = tid; i < NA; i += 256) {
        mx = fminf(mx, xt[i * 2]);
        my = fminf(my, xt[i * 2 + 1]);
    }
    sx[tid] = mx; sy[tid] = my;
    __syncthreads();
    for (int s = 128; s > 0; s >>= 1) {
        if (tid < s) { sx[tid] = fminf(sx[tid], sx[tid + s]); sy[tid] = fminf(sy[tid], sy[tid + s]); }
        __syncthreads();
    }
    float ltx = sx[0] - 1.2f, lty = sy[0] - 1.2f;       // margin = 2*N_SIZE*CELL
    for (int i = tid; i < NA; i += 256) {
        int m  = (int)masks[t * NA + i];
        int px = (int)floorf((xt[i * 2]     - ltx) / CELLSZ) * m;
        int py = (int)floorf((xt[i * 2 + 1] - lty) / CELLSZ) * m;
        cells[t * NA + i] = make_int2(px, py);
    }
}

// ---------------- fully fused timestep: social -> e-GEMM -> g-GEMM -> gates -> out ------------
// One block owns 16 agents. h is ping-ponged (h_in read by all blocks, h_out written per block).
__global__ __launch_bounds__(256)
void k_step(const int2* __restrict__ cells_t,
            const float* __restrict__ h_in, float* __restrict__ h_out,
            float* __restrict__ c_st,
            const float* __restrict__ masks_t, const float* __restrict__ xin,
            const float* __restrict__ W_in, const float* __restrict__ b_in,
            const f16_t* __restrict__ Wsoc_h, const float* __restrict__ b_soc,
            const f16_t* __restrict__ Wg_h, const float* __restrict__ b_g,
            const float* __restrict__ W_out, const float* __restrict__ b_out,
            float* __restrict__ out_t) {
    __shared__ int2  sCell[NA];            // 16KB: full per-frame cell table
    __shared__ f16_t sZh[AGB * ZK];        // 8KB : fused activation rows r|e|h
    __shared__ float sG[AGB * G4];         // 16KB: gate pre-activations
    __shared__ f16_t sAsoc[AGB * HID];     // 2KB : masked cell_val (A of e-GEMM)
    __shared__ float sH[AGB * HID];        // 4KB : new h for output matmul

    const int tid    = threadIdx.x;
    const int lane   = tid & 31;
    const int wv     = tid >> 5;           // 8 waves
    const int agent0 = blockIdx.x * AGB;

    // L2 prefetch of the fused gate weight (global_prefetch_b8 path)
    __builtin_prefetch(Wg_h + (size_t)tid * 256, 0, 1);

    // ---- stage cell table into LDS via async global->LDS DMA (ASYNCcnt path) ----
#pragma unroll
    for (int r = 0; r < NA / 256; ++r) {
        int idx = r * 256 + tid;
        unsigned ldsoff = (unsigned)(size_t)(void*)&sCell[idx];
        const int2* gp = cells_t + idx;
        asm volatile("global_load_async_to_lds_b64 %0, %1, off"
                     :: "v"(ldsoff), "v"(gp) : "memory");
    }
    asm volatile("s_wait_asynccnt 0x0" ::: "memory");
    __syncthreads();

    // ---- phase 1: social gather (per-lane scan + ballot), r embed, h -> f16 ----
    {
        int la0 = 2 * wv, la1 = 2 * wv + 1;            // two agents per wave
        int2 q0 = sCell[agent0 + la0]; q0.x -= 1; q0.y -= 1;
        int2 q1 = sCell[agent0 + la1]; q1.x -= 1; q1.y -= 1;
        float a00 = 0.f, a01 = 0.f, a10 = 0.f, a11 = 0.f;
        for (int base = 0; base < NA; base += 32) {
            int2 pj = sCell[base + lane];
            unsigned m0 = (unsigned)__ballot(pj.x == q0.x && pj.y == q0.y);
            unsigned m1 = (unsigned)__ballot(pj.x == q1.x && pj.y == q1.y);
            while (m0) {
                int b = __builtin_ctz(m0); m0 &= m0 - 1;
                const float* hj = h_in + (size_t)(base + b) * HID;
                a00 += hj[lane]; a01 += hj[lane + 32];
            }
            while (m1) {
                int b = __builtin_ctz(m1); m1 &= m1 - 1;
                const float* hj = h_in + (size_t)(base + b) * HID;
                a10 += hj[lane]; a11 += hj[lane + 32];
            }
        }
#pragma unroll
        for (int s = 0; s < 2; ++s) {
            int la = (s == 0) ? la0 : la1;
            int ag = agent0 + la;
            float m = masks_t[ag];                      // Hf = tile(cell_val)*mask
            float v0 = (s == 0) ? a00 : a10, v1 = (s == 0) ? a01 : a11;
            sAsoc[la * HID + lane]      = (f16_t)(v0 * m);
            sAsoc[la * HID + 32 + lane] = (f16_t)(v1 * m);
            // r = relu(x @ W_in.T + b_in), K=2 on VALU
            float x0 = xin[ag * 2], x1 = xin[ag * 2 + 1];
#pragma unroll
            for (int jj = 0; jj < 4; ++jj) {
                int idx = lane + jj * 32;
                float r = b_in[idx] + x0 * W_in[idx * 2] + x1 * W_in[idx * 2 + 1];
                sZh[la * ZK + idx] = (f16_t)fmaxf(r, 0.f);
            }
            sZh[la * ZK + 192 + lane] = (f16_t)h_in[(size_t)ag * HID + lane];
            sZh[la * ZK + 224 + lane] = (f16_t)h_in[(size_t)ag * HID + 32 + lane];
        }
    }
    __syncthreads();

    // ---- phase 2: e = relu(Asoc @ Wsoc_eff.T + b_soc) -> sZh[:,128:192] (WMMA) ----
    if (wv < 4) {                                       // 4 column tiles of 16
        int col0 = wv * 16;
        v8f acc = {};
#pragma unroll
        for (int k0 = 0; k0 < 64; k0 += 32) {
            v16h a = load_a_frag(sAsoc, HID, 0, k0, lane);
            v16h b = load_b_frag(Wsoc_h, HID, col0, k0, lane);
            acc = __builtin_amdgcn_wmma_f32_16x16x32_f16(false, a, false, b, (short)0, acc, false, false);
        }
        int n = lane & 15;
        float bias = b_soc[col0 + n];
        int rbase  = (lane >> 4) << 3;
#pragma unroll
        for (int v = 0; v < 8; ++v)
            sZh[(rbase + v) * ZK + MED + col0 + n] = (f16_t)fmaxf(acc[v] + bias, 0.f);
    }
    __syncthreads();

    // ---- phase 3: g = zh @ [Wih|Whh].T + b_g  (K=256, WMMA; 2 col tiles per wave) ----
    {
        int c0a = wv * 16, c0b = (wv + 8) * 16;
        v8f accA = {}, accB = {};
#pragma unroll
        for (int k0 = 0; k0 < ZK; k0 += 32) {
            v16h a  = load_a_frag(sZh, ZK, 0, k0, lane);
            v16h bA = load_b_frag(Wg_h, ZK, c0a, k0, lane);
            v16h bB = load_b_frag(Wg_h, ZK, c0b, k0, lane);
            accA = __builtin_amdgcn_wmma_f32_16x16x32_f16(false, a, false, bA, (short)0, accA, false, false);
            accB = __builtin_amdgcn_wmma_f32_16x16x32_f16(false, a, false, bB, (short)0, accB, false, false);
        }
        int n = lane & 15;
        int rbase = (lane >> 4) << 3;
        float biasA = b_g[c0a + n], biasB = b_g[c0b + n];
#pragma unroll
        for (int v = 0; v < 8; ++v) {
            sG[(rbase + v) * G4 + c0a + n] = accA[v] + biasA;
            sG[(rbase + v) * G4 + c0b + n] = accB[v] + biasB;
        }
    }
    __syncthreads();

    // ---- phase 4: LSTM gates; update c (global), h (ping-pong out), sH ----
#pragma unroll
    for (int r = 0; r < (AGB * HID) / 256; ++r) {       // 4 items per thread
        int item = r * 256 + tid;
        int a = item >> 6, u = item & 63;
        const float* gr = sG + a * G4;
        float gi = gr[u], gf = gr[64 + u], gc = gr[128 + u], go = gr[192 + u];
        float si = 1.f / (1.f + expf(-gi));
        float sf = 1.f / (1.f + expf(-gf));
        float so = 1.f / (1.f + expf(-go));
        size_t gidx = (size_t)(agent0 + a) * HID + u;
        float cn = sf * c_st[gidx] + si * tanhf(gc);
        float hn = so * tanhf(cn);
        c_st[gidx]  = cn;
        h_out[gidx] = hn;
        sH[item]    = hn;
    }
    __syncthreads();

    // ---- phase 5: out = (h_new @ W_out.T + b_out) * mask ----
    if (tid < AGB * 2) {
        int a = tid >> 1, d = tid & 1;
        const float* hn = sH + a * HID;
        float acc = 0.f;
#pragma unroll 8
        for (int u = 0; u < HID; ++u) acc += hn[u] * W_out[d * 64 + u];
        int ag = agent0 + a;
        out_t[ag * 2 + d] = (acc + b_out[d]) * masks_t[ag];
    }
}

// ---------------- host orchestration ----------------
extern "C" void kernel_launch(void* const* d_in, const int* in_sizes, int n_in,
                              void* d_out, int out_size, void* d_ws, size_t ws_size,
                              hipStream_t stream) {
    (void)in_sizes; (void)n_in; (void)out_size; (void)ws_size;
    const float* X     = (const float*)d_in[0];
    const float* masks = (const float*)d_in[1];
    const float* h0    = (const float*)d_in[2];
    const float* c0    = (const float*)d_in[3];
    // d_in[4] = Y (unused), d_in[5]/d_in[6] = T_obs/T_pred (constants 9/19 per reference)
    const float* W_in  = (const float*)d_in[7];
    const float* b_in  = (const float*)d_in[8];
    const float* W_soc = (const float*)d_in[9];
    const float* b_soc = (const float*)d_in[10];
    const float* W_ih  = (const float*)d_in[11];
    const float* W_hh  = (const float*)d_in[12];
    const float* b_ih  = (const float*)d_in[13];
    const float* b_hh  = (const float*)d_in[14];
    const float* W_out = (const float*)d_in[15];
    const float* b_out = (const float*)d_in[16];
    float* out = (float*)d_out;

    char* ws = (char*)d_ws;                              // all offsets 256B aligned
    f16_t* Wg_h   = (f16_t*)(ws + 0);                    // 256*256*2 = 131072
    f16_t* Wsoc_h = (f16_t*)(ws + 131072);               // 64*64*2   = 8192
    float* b_g    = (float*)(ws + 139264);               // 256*4     = 1024
    int2*  cells  = (int2*)(ws + 140288);                // 20*2048*8 = 327680
    float* hA     = (float*)(ws + 467968);               // 2048*64*4 = 524288
    float* hB     = (float*)(ws + 992256);               // 524288
    float* cbuf   = (float*)(ws + 1516544);              // 524288   (end ~2MB)

    k_prep <<<512, 256, 0, stream>>>(W_ih, W_hh, W_soc, b_ih, b_hh, h0, c0,
                                     Wg_h, Wsoc_h, b_g, hA, cbuf, out);
    k_cells<<<20, 256, 0, stream>>>(X, masks, cells);

    for (int t = 0; t <= T_PRED; ++t) {
        // pred frames (t>=10): r comes from last_pt = X[10] at t==10, else out[t-2]
        const float* xin = (t <= T_OBS) ? (X + (size_t)t * NA * 2)
                         : (t == T_OBS + 1) ? (X + (size_t)(T_OBS + 1) * NA * 2)
                                            : (out + (size_t)(t - 2) * NA * 2);
        const float* hi = (t & 1) ? hB : hA;
        float*       ho = (t & 1) ? hA : hB;
        k_step<<<NA / AGB, 256, 0, stream>>>(cells + t * NA, hi, ho, cbuf,
                                             masks + t * NA, xin, W_in, b_in,
                                             Wsoc_h, b_soc, Wg_h, b_g,
                                             W_out, b_out, out + (size_t)t * NA * 2);
    }
}
